// DeformableTransformerDecoderLayer_11175504904379
// MI455X (gfx1250) — matrix operand order repaired and checked by reading, not compiled, not run on hardware
//
#include <hip/hip_runtime.h>
#include <hip/hip_bf16.h>

// ---------------------------------------------------------------------------
// Deformable Transformer Decoder Layer for MI455X (gfx1250, wave32, WMMA)
// D=256, DFF=1024, NL=4, NH=8, NP=4, NQ=900, BS=8, DH=32, LEN_IN=13294
// ---------------------------------------------------------------------------

typedef __attribute__((ext_vector_type(16))) _Float16 v16h;
typedef __attribute__((ext_vector_type(8)))  float    v8f;

#define NQc 900
#define BSc 8
#define Dc 256
#define DHc 32
#define NHc 8
#define NLc 4
#define NPc 4
#define DFFc 1024
#define MROWS (NQc * BSc)            // 7200
#define LENIN 13294
#define VROWS (LENIN * BSc)          // 106352

// ---------------- WMMA fragment loaders (contiguous k order, A and B alike) --------------
// A 16x32 (MxK), row-major LDS tile with row stride 32 halfs.
// lane<16: M=lane,   K elems = {c0..c0+15} with c0=0
// lane>=16: M=lane-16, same with c0=16
__device__ inline v16h load_a_16x32(const _Float16* __restrict__ tile, int lane) {
    const int m  = lane & 15;
    const int c0 = (lane < 16) ? 0 : 16;
    const _Float16* p = tile + m * 32;
    v16h a;
#pragma unroll
    for (int e = 0; e < 8; ++e) a[e] = p[c0 + e];
#pragma unroll
    for (int e = 0; e < 8; ++e) a[8 + e] = p[c0 + 8 + e];
    return a;
}

// B 32x16 (KxN) sourced from an N-major tile: tile[n*32 + k].
// lane: N = lane&15, K elems = koff..koff+15 with koff = (lane<16)?0:16
__device__ inline v16h load_b_32x16(const _Float16* __restrict__ tileN, int lane) {
    const int n    = lane & 15;
    const int koff = (lane < 16) ? 0 : 16;
    const _Float16* p = tileN + n * 32 + koff;
    v16h b;
#pragma unroll
    for (int e = 0; e < 16; ++e) b[e] = p[e];
    return b;
}

// ---------------------------- generic GEMM ----------------------------------
// C[M,N] = A[M,K] @ W[N,K]^T + bias[N]   (optional ReLU, OutT = float/_Float16)
// Block tile 128(M) x 64(N); 8 waves, each wave owns a 16(M) x 64(N) strip so
// one A fragment feeds 4 WMMAs (A reuse; 1.25 LDS fragment loads per WMMA).
template <bool RELU, typename OutT>
__global__ __launch_bounds__(256) void gemm_bias_kernel(
    const float* __restrict__ A, const float* __restrict__ W,
    const float* __restrict__ bias, OutT* __restrict__ C,
    int M, int N, int K) {
    __shared__ _Float16 As[128 * 32];
    __shared__ _Float16 Bs[64 * 32];

    const int m0   = blockIdx.x * 128;
    const int n0   = blockIdx.y * 64;
    const int t    = threadIdx.x;
    const int lane = t & 31;
    const int wave = t >> 5;     // 0..7 -> 16-row M strip

    v8f acc[4] = {v8f{}, v8f{}, v8f{}, v8f{}};

    for (int k0 = 0; k0 < K; k0 += 32) {
        __syncthreads();
#pragma unroll
        for (int i = 0; i < 16; ++i) {           // A tile 128x32
            int idx = t + i * 256;
            int r = idx >> 5, c = idx & 31;
            int gr = m0 + r;
            float v = (gr < M) ? A[(size_t)gr * K + k0 + c] : 0.0f;
            As[r * 32 + c] = (_Float16)v;
        }
#pragma unroll
        for (int i = 0; i < 8; ++i) {            // W tile 64x32 (N-major)
            int idx = t + i * 256;
            int r = idx >> 5, c = idx & 31;
            int gn = n0 + r;
            float v = (gn < N) ? W[(size_t)gn * K + k0 + c] : 0.0f;
            Bs[r * 32 + c] = (_Float16)v;
        }
        // prefetch next K-tile rows while this tile computes (global_prefetch_b8)
        if (k0 + 32 < K) {
            if (t < 128) {
                int gr = m0 + t;
                if (gr < M) __builtin_prefetch(&A[(size_t)gr * K + k0 + 32], 0, 0);
            } else if (t < 192) {
                int gn = n0 + (t - 128);
                if (gn < N) __builtin_prefetch(&W[(size_t)gn * K + k0 + 32], 0, 0);
            }
        }
        __syncthreads();

        v16h a = load_a_16x32(&As[(wave * 16) * 32], lane);
#pragma unroll
        for (int g = 0; g < 4; ++g) {
            v16h b = load_b_32x16(&Bs[(g * 16) * 32], lane);
            acc[g] = __builtin_amdgcn_wmma_f32_16x16x32_f16(false, a, false, b,
                                                            (short)0, acc[g],
                                                            false, false);
        }
    }

    const int mbase = m0 + wave * 16 + ((lane < 16) ? 0 : 8);
    const int col0  = n0 + (lane & 15);
#pragma unroll
    for (int r = 0; r < 8; ++r) {
        int row = mbase + r;
        if (row < M) {
#pragma unroll
            for (int g = 0; g < 4; ++g) {
                float v = acc[g][r] + bias[col0 + g * 16];
                if (RELU) v = fmaxf(v, 0.0f);
                C[(size_t)row * N + col0 + g * 16] = (OutT)v;
            }
        }
    }
}

// ------------------------- elementwise add ----------------------------------
__global__ __launch_bounds__(256) void add_kernel(const float* __restrict__ a,
                                                  const float* __restrict__ b,
                                                  float* __restrict__ c, int n) {
    int i = blockIdx.x * 256 + threadIdx.x;
    if (i < n) c[i] = a[i] + b[i];
}

// --------------------- residual add + LayerNorm (N=256) ---------------------
__global__ __launch_bounds__(256) void add_ln_kernel(
    const float* __restrict__ X, const float* __restrict__ R,
    const float* __restrict__ g, const float* __restrict__ bt,
    float* __restrict__ O, int M) {
    int row  = blockIdx.x * 8 + (threadIdx.x >> 5);
    int lane = threadIdx.x & 31;
    if (row >= M) return;
    const float* x = X + (size_t)row * Dc;
    const float* r = R + (size_t)row * Dc;
    float v[8];
    float s = 0.0f;
#pragma unroll
    for (int i = 0; i < 8; ++i) { v[i] = x[lane + i * 32] + r[lane + i * 32]; s += v[i]; }
#pragma unroll
    for (int off = 1; off < 32; off <<= 1) s += __shfl_xor(s, off, 32);
    float mean = s * (1.0f / Dc);
    float var = 0.0f;
#pragma unroll
    for (int i = 0; i < 8; ++i) { float d = v[i] - mean; var += d * d; }
#pragma unroll
    for (int off = 1; off < 32; off <<= 1) var += __shfl_xor(var, off, 32);
    var *= (1.0f / Dc);
    float inv = rsqrtf(var + 1e-5f);
    float* o = O + (size_t)row * Dc;
#pragma unroll
    for (int i = 0; i < 8; ++i) {
        int c = lane + i * 32;
        o[c] = (v[i] - mean) * inv * g[c] + bt[c];
    }
}

// ------------------- flash self-attention (1 wave / block) ------------------
// QK: [MROWS, 512] (q cols 0..255, k cols 256..511), V: [MROWS, 256]
// SA out: [MROWS, 256].  grid = (57, NH, BS), block = 32.
__global__ __launch_bounds__(32) void attn_kernel(
    const float* __restrict__ QK, const float* __restrict__ V,
    float* __restrict__ SA) {
    __shared__ _Float16 Pt[16 * 32];   // P tile (16q x 32k)
    __shared__ _Float16 Vt[32 * 32];   // V^T tile (32d x 32k)

    const int qt = blockIdx.x, h = blockIdx.y, b = blockIdx.z;
    const int lane = threadIdx.x;
    const int c0   = (lane < 16) ? 0 : 16;
    const float scale = 0.17677669529663687f;  // 1/sqrt(32)

    // Q fragment (persistent across key loop)
    v16h qa;
    {
        int qi = qt * 16 + (lane & 15);
        if (qi > NQc - 1) qi = NQc - 1;
        const float* qrow = QK + ((size_t)qi * BSc + b) * 512 + h * DHc;
#pragma unroll
        for (int e = 0; e < 8; ++e) qa[e] = (_Float16)qrow[c0 + e];
#pragma unroll
        for (int e = 0; e < 8; ++e) qa[8 + e] = (_Float16)qrow[c0 + 8 + e];
    }

    v8f acc0 = {}, acc1 = {};
    float mrun[8], lrun[8];
#pragma unroll
    for (int r = 0; r < 8; ++r) { mrun[r] = -1e30f; lrun[r] = 0.0f; }

    for (int kb = 0; kb < NQc; kb += 32) {
        // K^T fragments (B-layout: contiguous DH run per lane half)
        v16h kf0, kf1;
        {
            int key = kb + (lane & 15);
            if (key > NQc - 1) key = NQc - 1;
            const float* kr = QK + ((size_t)key * BSc + b) * 512 + 256 + h * DHc +
                              ((lane < 16) ? 0 : 16);
#pragma unroll
            for (int e = 0; e < 16; ++e) kf0[e] = (_Float16)kr[e];
        }
        {
            int key = kb + 16 + (lane & 15);
            if (key > NQc - 1) key = NQc - 1;
            const float* kr = QK + ((size_t)key * BSc + b) * 512 + 256 + h * DHc +
                              ((lane < 16) ? 0 : 16);
#pragma unroll
            for (int e = 0; e < 16; ++e) kf1[e] = (_Float16)kr[e];
        }

        v8f zero = {};
        v8f s0 = __builtin_amdgcn_wmma_f32_16x16x32_f16(false, qa, false, kf0,
                                                        (short)0, zero, false, false);
        v8f s1 = __builtin_amdgcn_wmma_f32_16x16x32_f16(false, qa, false, kf1,
                                                        (short)0, zero, false, false);

        const bool ok0 = (kb + (lane & 15)) < NQc;
        const bool ok1 = (kb + 16 + (lane & 15)) < NQc;

        // streaming softmax per row (8 rows per lane half)
#pragma unroll
        for (int r = 0; r < 8; ++r) {
            float a0 = ok0 ? s0[r] * scale : -1e30f;
            float a1 = ok1 ? s1[r] * scale : -1e30f;
            float mx = fmaxf(a0, a1);
            mx = fmaxf(mx, __shfl_xor(mx, 1, 16));
            mx = fmaxf(mx, __shfl_xor(mx, 2, 16));
            mx = fmaxf(mx, __shfl_xor(mx, 4, 16));
            mx = fmaxf(mx, __shfl_xor(mx, 8, 16));
            float mnew = fmaxf(mrun[r], mx);
            float corr = __expf(mrun[r] - mnew);
            float p0 = __expf(a0 - mnew);
            float p1 = __expf(a1 - mnew);
            float ps = p0 + p1;
            ps += __shfl_xor(ps, 1, 16);
            ps += __shfl_xor(ps, 2, 16);
            ps += __shfl_xor(ps, 4, 16);
            ps += __shfl_xor(ps, 8, 16);
            lrun[r] = lrun[r] * corr + ps;
            mrun[r] = mnew;
            acc0[r] *= corr;
            acc1[r] *= corr;
            int prow = r + ((lane < 16) ? 0 : 8);
            Pt[prow * 32 + (lane & 15)]      = (_Float16)p0;
            Pt[prow * 32 + 16 + (lane & 15)] = (_Float16)p1;
        }

        // V tile -> LDS transposed (Vt[d][key])
        {
            int key = kb + lane;
            bool ok = key < NQc;
            if (key > NQc - 1) key = NQc - 1;
            const float* vr = V + ((size_t)key * BSc + b) * Dc + h * DHc;
#pragma unroll
            for (int d = 0; d < DHc; ++d)
                Vt[d * 32 + lane] = (_Float16)(ok ? vr[d] : 0.0f);
        }
        __syncthreads();

        v16h pa  = load_a_16x32(Pt, lane);
        v16h vb0 = load_b_32x16(&Vt[0], lane);         // d = 0..15
        v16h vb1 = load_b_32x16(&Vt[16 * 32], lane);   // d = 16..31
        acc0 = __builtin_amdgcn_wmma_f32_16x16x32_f16(false, pa, false, vb0,
                                                      (short)0, acc0, false, false);
        acc1 = __builtin_amdgcn_wmma_f32_16x16x32_f16(false, pa, false, vb1,
                                                      (short)0, acc1, false, false);
        __syncthreads();
    }

#pragma unroll
    for (int r = 0; r < 8; ++r) {
        int qi = qt * 16 + r + ((lane < 16) ? 0 : 8);
        if (qi < NQc) {
            float inv = 1.0f / lrun[r];
            float* o = SA + ((size_t)qi * BSc + b) * Dc + h * DHc;
            o[(lane & 15)]      = acc0[r] * inv;
            o[16 + (lane & 15)] = acc1[r] * inv;
        }
    }
}

// --------------- deformable sampling (1 wave per (q,b,h)) -------------------
// value: f16 [VROWS, 256], row = (start+pix)*BS + b
__global__ __launch_bounds__(256) void deform_sample_kernel(
    const _Float16* __restrict__ value, const float* __restrict__ offb,
    const float* __restrict__ awb, const float* __restrict__ ref,
    float* __restrict__ ca) {
    const int gw   = blockIdx.x * 8 + (threadIdx.x >> 5);
    const int lane = threadIdx.x & 31;
    if (gw >= NQc * BSc * NHc) return;
    const int h = gw & 7, b = (gw >> 3) & 7, q = gw >> 6;
    const int rowqb = q * BSc + b;

    const int Hs[4]  = {100, 50, 25, 13};
    const int Ws[4]  = {100, 50, 25, 13};
    const int St[4]  = {0, 10000, 12500, 13125};

    // softmax over the 16 (level,point) logits for this head (redundant/lane)
    const float* awp = awb + (size_t)rowqb * (NHc * NLc * NPc) + h * 16;
    float lg[16], mx = -1e30f;
#pragma unroll
    for (int j = 0; j < 16; ++j) { lg[j] = awp[j]; mx = fmaxf(mx, lg[j]); }
    float sum = 0.0f;
#pragma unroll
    for (int j = 0; j < 16; ++j) { lg[j] = __expf(lg[j] - mx); sum += lg[j]; }
    const float inv = 1.0f / sum;

    const float* offp = offb + (size_t)rowqb * Dc + h * DHc;
    float acc = 0.0f;

#pragma unroll
    for (int l = 0; l < NLc; ++l) {
        const int Hl = Hs[l], Wl = Ws[l], s0l = St[l];
        const float rx = ref[(((size_t)q * BSc + b) * NLc + l) * 2 + 0];
        const float ry = ref[(((size_t)q * BSc + b) * NLc + l) * 2 + 1];
#pragma unroll
        for (int p = 0; p < NPc; ++p) {
            const int j = l * NPc + p;
            const float ox = offp[j * 2 + 0];
            const float oy = offp[j * 2 + 1];
            const float x = rx * (float)Wl + ox - 0.5f;
            const float y = ry * (float)Hl + oy - 0.5f;
            const float fx = floorf(x), fy = floorf(y);
            const float dx = x - fx, dy = y - fy;
            const int x0 = (int)fx, y0 = (int)fy;
            const float w = lg[j] * inv;

            auto corner = [&](int yy, int xx, float wgt) {
                bool valid = (yy >= 0) && (yy < Hl) && (xx >= 0) && (xx < Wl);
                int yc = min(max(yy, 0), Hl - 1);
                int xc = min(max(xx, 0), Wl - 1);
                size_t vrow = (size_t)(s0l + yc * Wl + xc) * BSc + b;
                float val = (float)value[vrow * Dc + h * DHc + lane];
                acc += valid ? (w * wgt * val) : 0.0f;
            };
            corner(y0,     x0,     (1.0f - dy) * (1.0f - dx));
            corner(y0,     x0 + 1, (1.0f - dy) * dx);
            corner(y0 + 1, x0,     dy * (1.0f - dx));
            corner(y0 + 1, x0 + 1, dy * dx);
        }
    }
    ca[(size_t)rowqb * Dc + h * DHc + lane] = acc;
}

// ---------------------------------------------------------------------------
extern "C" void kernel_launch(void* const* d_in, const int* in_sizes, int n_in,
                              void* d_out, int out_size, void* d_ws, size_t ws_size,
                              hipStream_t stream) {
    (void)in_sizes; (void)n_in; (void)out_size; (void)ws_size;
    const float* tgt     = (const float*)d_in[0];
    const float* qpos    = (const float*)d_in[1];
    const float* ref     = (const float*)d_in[2];
    const float* memory  = (const float*)d_in[3];
    const float* sa_in_w = (const float*)d_in[6];
    const float* sa_in_b = (const float*)d_in[7];
    const float* sa_ow   = (const float*)d_in[8];
    const float* sa_ob   = (const float*)d_in[9];
    const float* off_w   = (const float*)d_in[10];
    const float* off_b   = (const float*)d_in[11];
    const float* aw_w    = (const float*)d_in[12];
    const float* aw_b    = (const float*)d_in[13];
    const float* val_w   = (const float*)d_in[14];
    const float* val_b   = (const float*)d_in[15];
    const float* out_w   = (const float*)d_in[16];
    const float* out_b   = (const float*)d_in[17];
    const float* ln1_g   = (const float*)d_in[18];
    const float* ln1_b   = (const float*)d_in[19];
    const float* ln2_g   = (const float*)d_in[20];
    const float* ln2_b   = (const float*)d_in[21];
    const float* ln3_g   = (const float*)d_in[22];
    const float* ln3_b   = (const float*)d_in[23];
    const float* ffn1_w  = (const float*)d_in[24];
    const float* ffn1_b  = (const float*)d_in[25];
    const float* ffn2_w  = (const float*)d_in[26];
    const float* ffn2_b  = (const float*)d_in[27];
    float* out = (float*)d_out;

    // ------------------ workspace layout (with aliasing) -------------------
    char* ws = (char*)d_ws;
    size_t off = 0;
    auto alloc = [&](size_t bytes) {
        void* p = ws + off;
        off += (bytes + 255) & ~(size_t)255;
        return p;
    };
    float* f_qk   = (float*)alloc((size_t)MROWS * Dc * 4);      // qk, later query
    float* f_qkv  = (float*)alloc((size_t)MROWS * DFFc * 4);    // Q|K (512), later ffn hidden (1024)
    float* f_v    = (float*)alloc((size_t)MROWS * Dc * 4);      // v, later ca
    float* f_sa   = (float*)alloc((size_t)MROWS * Dc * 4);      // sa, later ca2
    float* f_sa2  = (float*)alloc((size_t)MROWS * Dc * 4);      // sa-proj, later offsets
    float* f_t2   = (float*)alloc((size_t)MROWS * Dc * 4);      // tgt after LN2
    _Float16* f_val = (_Float16*)alloc((size_t)VROWS * Dc * 2); // projected value (f16)
    float* f_aw   = (float*)alloc((size_t)MROWS * 128 * 4);     // attn-weight logits
    float* f_t3   = (float*)alloc((size_t)MROWS * Dc * 4);      // tgt after LN1
    float* f_ff2  = (float*)alloc((size_t)MROWS * Dc * 4);      // ffn output

    const int nQD = MROWS * Dc;
    const int MB  = (MROWS + 127) / 128;   // 57
    const int VB  = (VROWS + 127) / 128;   // 831

    // 1) qk = tgt + qpos
    add_kernel<<<(nQD + 255) / 256, 256, 0, stream>>>(tgt, qpos, f_qk, nQD);
    // 2) [Q|K] = qk @ sa_in_w[0:512].T + b[0:512]
    gemm_bias_kernel<false, float><<<dim3(MB, 8), 256, 0, stream>>>(
        f_qk, sa_in_w, sa_in_b, f_qkv, MROWS, 512, Dc);
    // 3) V = tgt @ sa_in_w[512:768].T + b[512:768]
    gemm_bias_kernel<false, float><<<dim3(MB, 4), 256, 0, stream>>>(
        tgt, sa_in_w + (size_t)512 * Dc, sa_in_b + 512, f_v, MROWS, Dc, Dc);
    // 4) flash self-attention
    attn_kernel<<<dim3(57, NHc, BSc), 32, 0, stream>>>(f_qkv, f_v, f_sa);
    // 5) sa2 = sa @ sa_out_w.T + b
    gemm_bias_kernel<false, float><<<dim3(MB, 4), 256, 0, stream>>>(
        f_sa, sa_ow, sa_ob, f_sa2, MROWS, Dc, Dc);
    // 6) tgt2 = LN2(tgt + sa2)
    add_ln_kernel<<<(MROWS + 7) / 8, 256, 0, stream>>>(tgt, f_sa2, ln2_g, ln2_b,
                                                       f_t2, MROWS);
    // 7) query = tgt2 + qpos  (reuses f_qk)
    add_kernel<<<(nQD + 255) / 256, 256, 0, stream>>>(f_t2, qpos, f_qk, nQD);
    // 8) value = memory @ val_w.T + val_b   (f16 out)
    gemm_bias_kernel<false, _Float16><<<dim3(VB, 4), 256, 0, stream>>>(
        memory, val_w, val_b, f_val, VROWS, Dc, Dc);
    // 9) offsets = query @ off_w.T + off_b  (reuses f_sa2)
    gemm_bias_kernel<false, float><<<dim3(MB, 4), 256, 0, stream>>>(
        f_qk, off_w, off_b, f_sa2, MROWS, Dc, Dc);
    // 10) aw logits = query @ aw_w.T + aw_b
    gemm_bias_kernel<false, float><<<dim3(MB, 2), 256, 0, stream>>>(
        f_qk, aw_w, aw_b, f_aw, MROWS, 128, Dc);
    // 11) deformable bilinear sampling -> ca (reuses f_v)
    deform_sample_kernel<<<(NQc * BSc * NHc) / 8, 256, 0, stream>>>(
        f_val, f_sa2, f_aw, ref, f_v);
    // 12) ca2 = ca @ out_w.T + out_b (reuses f_sa)
    gemm_bias_kernel<false, float><<<dim3(MB, 4), 256, 0, stream>>>(
        f_v, out_w, out_b, f_sa, MROWS, Dc, Dc);
    // 13) tgt3 = LN1(tgt2 + ca2)
    add_ln_kernel<<<(MROWS + 7) / 8, 256, 0, stream>>>(f_t2, f_sa, ln1_g, ln1_b,
                                                       f_t3, MROWS);
    // 14) h = relu(tgt3 @ ffn1_w.T + b1)  (reuses f_qkv region, 1024 wide)
    gemm_bias_kernel<true, float><<<dim3(MB, 16), 256, 0, stream>>>(
        f_t3, ffn1_w, ffn1_b, f_qkv, MROWS, DFFc, Dc);
    // 15) ff2 = h @ ffn2_w.T + b2
    gemm_bias_kernel<false, float><<<dim3(MB, 4), 256, 0, stream>>>(
        f_qkv, ffn2_w, ffn2_b, f_ff2, MROWS, Dc, DFFc);
    // 16) out = LN3(tgt3 + ff2)
    add_ln_kernel<<<(MROWS + 7) / 8, 256, 0, stream>>>(f_t3, f_ff2, ln3_g, ln3_b,
                                                       out, MROWS);
}
